// LSTMcell_50955492000337
// MI455X (gfx1250) — matrix operand order repaired and checked by reading, not compiled, not run on hardware
//
#include <hip/hip_runtime.h>

typedef __attribute__((ext_vector_type(16))) __bf16 v16bf;
typedef __attribute__((ext_vector_type(8)))  float  v8f;

#define BSZ    64
#define HDIM   512
#define GDIM   2048      // 4*H
#define TSTEPS 256
#define MT     4         // 64/16 m-tiles
#define NT     128       // 2048/16 n-tiles
#define KBLK   16        // 512/32 k-blocks

// ---- workspace layout (bytes) ----
#define OFF_UFRAG  ((size_t)0)
#define OFF_VFRAG  (OFF_UFRAG + (size_t)GDIM*HDIM*2)            // 2 MB
#define OFF_XFRAG  (OFF_VFRAG + (size_t)GDIM*HDIM*2)            // 2 MB
#define OFF_GATESX (OFF_XFRAG + (size_t)TSTEPS*BSZ*HDIM*2)      // 16 MB
#define OFF_HFRAG  (OFF_GATESX + (size_t)TSTEPS*BSZ*GDIM*4)     // 134 MB, 2 ping-pong h buffers
#define OFF_BAR    (OFF_HFRAG + (size_t)2*BSZ*HDIM*2)

#define HFRAG_ELEMS ((size_t)BSZ*HDIM)   // 32768 bf16 per buffer

__device__ __forceinline__ float fast_sigmoid(float x) {
    return 1.0f / (1.0f + __expf(-x));
}
__device__ __forceinline__ float fast_tanh(float x) {
    float e = __expf(-2.0f * fabsf(x));
    float r = (1.0f - e) / (1.0f + e);
    return copysignf(r, x);
}

// Pack weight W (H x 4H, row-major, f32) into bf16 B-fragment order:
// frag[((n_tile*16+kb)*32 + lane)*16 + h], col=lane&15, K=(h&7)+((h>>3)<<4)+((lane>>4)<<3)
__global__ void pack_w(const float* __restrict__ W, __bf16* __restrict__ frag) {
    int d      = blockIdx.x * 256 + threadIdx.x;   // GDIM*HDIM elements
    int h      = d & 15;
    int lane   = (d >> 4) & 31;
    int kb     = (d >> 9) & 15;
    int n_tile = d >> 13;
    int n = n_tile * 16 + (lane & 15);
    int k = kb * 32 + (h & 7) + ((h >> 3) << 4) + ((lane >> 4) << 3);
    frag[d] = (__bf16)W[(size_t)k * GDIM + n];
}

// Pack x (BS, H, T) f32 into bf16 A-fragment order per 16-bit 16x32 A layout.
__global__ void pack_x(const float* __restrict__ x, __bf16* __restrict__ frag) {
    int d      = blockIdx.x * 256 + threadIdx.x;   // T*BS*H elements
    int h      = d & 15;
    int lane   = (d >> 4) & 31;
    int kb     = (d >> 9) & 15;
    int m_tile = (d >> 13) & 3;
    int t      = d >> 15;
    int M = lane & 15;
    int K = (h & 7) + ((h >> 3) << 4) + ((lane >> 4) << 3);
    int b = m_tile * 16 + M;
    int k = kb * 32 + K;
    frag[d] = (__bf16)x[((size_t)b * HDIM + k) * TSTEPS + t];
}

// Batched GEMM: gatesx[t] = x_t * U + bias, one 16x16 tile per wave, stored in C-fragment order.
__global__ void gemm_xu(const __bf16* __restrict__ xfrag, const __bf16* __restrict__ ufrag,
                        const float* __restrict__ bias, float* __restrict__ gatesx) {
    int wave = (blockIdx.x * blockDim.x + threadIdx.x) >> 5;  // tile id = t*512 + m_tile*128 + n_tile
    int lane = threadIdx.x & 31;
    int n_tile = wave & 127;
    int m_tile = (wave >> 7) & 3;
    int t      = wave >> 9;

    const v16bf* A = (const v16bf*)(xfrag + ((size_t)(t * MT + m_tile) * KBLK) * 32 * 16);
    const v16bf* B = (const v16bf*)(ufrag + ((size_t)n_tile * KBLK) * 32 * 16);

    v8f acc = {};
    for (int kb = 0; kb < KBLK; ++kb) {
        if (kb + 1 < KBLK) {
            __builtin_prefetch((const void*)&A[(kb + 1) * 32 + lane], 0, 0);
            __builtin_prefetch((const void*)&B[(kb + 1) * 32 + lane], 0, 0);
        }
        v16bf av = A[kb * 32 + lane];
        v16bf bv = B[kb * 32 + lane];
        acc = __builtin_amdgcn_wmma_f32_16x16x32_bf16(false, av, false, bv,
                                                      (short)0, acc, false, false);
    }
    float bb = bias[n_tile * 16 + (lane & 15)];
#pragma unroll
    for (int r = 0; r < 8; ++r) acc[r] += bb;
    *(v8f*)(gatesx + (size_t)wave * 256 + lane * 8) = acc;
}

__device__ __forceinline__ void grid_sync(unsigned* bar, unsigned nblk) {
    __threadfence();     // release this thread's global writes (h fragments)
    __syncthreads();
    if (threadIdx.x == 0) {
        volatile unsigned* vcnt = bar;
        volatile unsigned* vgen = bar + 1;
        unsigned g = *vgen;
        if (atomicAdd(bar, 1u) == nblk - 1) {
            *vcnt = 0;
            __threadfence();
            atomicAdd(bar + 1, 1u);
        } else {
            while (*vgen == g) __builtin_amdgcn_s_sleep(1);
        }
        __threadfence();
    }
    __syncthreads();
}

// Persistent recurrent kernel: 64 blocks x 8 waves.
// Block (jt, m_group) owns all FOUR gate tiles (i,f,g,o) for hidden column tile jt
// and batch rows [m_group*32, m_group*32+32). Gates never leave LDS; cell state c
// never leaves registers; one grid barrier per timestep (h broadcast, ping-pong).
__global__ void lstm_recurrent(const __bf16* __restrict__ vfrag,
                               const float*  __restrict__ gatesx,
                               __bf16*       __restrict__ hfrag,   // 2 ping-pong buffers
                               float*        __restrict__ out,
                               unsigned*     __restrict__ bar) {
    extern __shared__ char lds_raw[];
    __bf16* ldsV = (__bf16*)lds_raw;              // 4 gate tiles * 8192 bf16 = 64 KB
    float*  ldsG = (float*)(lds_raw + 65536);     // 8 waves * 256 f32       =  8 KB

    const int tid     = threadIdx.x;
    const int lane    = tid & 31;
    const int wave    = tid >> 5;                 // 0..7
    const int gate    = wave & 3;                 // i,f,g,o
    const int m_local = wave >> 2;                // 0..1
    const int jt      = blockIdx.x & 31;          // hidden column tile 0..31
    const int m_group = blockIdx.x >> 5;          // 0..1
    const int m_tile  = m_group * 2 + m_local;    // 0..3
    const int n_tile  = gate * 32 + jt;           // 0..127

    // one-time V stage-in: 4 gate tiles x 16 KB, resident for all 256 steps
    for (int g = 0; g < 4; ++g) {
        const uint4* src = (const uint4*)(vfrag + (size_t)(g * 32 + jt) * 8192);
        uint4* dst = (uint4*)(ldsV + g * 8192);
        for (int i = tid; i < 1024; i += 256) dst[i] = src[i];
    }
    __syncthreads();

    const v16bf* Blds = (const v16bf*)(ldsV + gate * 8192);

    // per-thread persistent cell state: 2 fixed cells for all timesteps
    float creg[2] = {0.0f, 0.0f};

    for (int t = 0; t < TSTEPS; ++t) {
        // start from precomputed x_t*U + bias (C-fragment order)
        v8f acc = *(const v8f*)(gatesx +
                   (((size_t)t * MT + m_tile) * NT + n_tile) * 256 + lane * 8);
        const v16bf* A = (const v16bf*)(hfrag + (size_t)(t & 1) * HFRAG_ELEMS +
                                        (size_t)m_tile * KBLK * 32 * 16);
        for (int kb = 0; kb < KBLK; ++kb) {
            v16bf av = A[kb * 32 + lane];
            v16bf bv = Blds[kb * 32 + lane];
            acc = __builtin_amdgcn_wmma_f32_16x16x32_bf16(false, av, false, bv,
                                                          (short)0, acc, false, false);
        }
        // stage gate tile in LDS (C-fragment order)
        *(v8f*)(ldsG + wave * 256 + lane * 8) = acc;
        __syncthreads();

        // fused cell update: 512 block-local cells over 256 threads
        __bf16* hnext = hfrag + (size_t)((t + 1) & 1) * HFRAG_ELEMS;
#pragma unroll
        for (int rep = 0; rep < 2; ++rep) {
            int lc   = tid + rep * 256;        // 0..511
            int r    = lc >> 4;                // row within block's 32 rows
            int cidx = lc & 15;                // col within 16 hidden columns
            int b    = m_group * 32 + r;
            int j    = jt * 16 + cidx;
            int mrow = r >> 4;                 // which m_local tile
            int M    = r & 15;
            int claneA = cidx + ((M >= 8) ? 16 : 0);
            int ridx   = M & 7;
            int cpos   = claneA * 8 + ridx;
            float gi = ldsG[(mrow * 4 + 0) * 256 + cpos];
            float gf = ldsG[(mrow * 4 + 1) * 256 + cpos];
            float gg = ldsG[(mrow * 4 + 2) * 256 + cpos];
            float go = ldsG[(mrow * 4 + 3) * 256 + cpos];
            float it = fast_sigmoid(gi);
            float ft = fast_sigmoid(gf);
            float gt = fast_tanh(gg);
            float ot = fast_sigmoid(go);
            float cn = ft * creg[rep] + it * gt;
            float hn = ot * fast_tanh(cn);
            creg[rep] = cn;
            out[((size_t)b * HDIM + j) * TSTEPS + t] = hn;   // hidden_seq[b][j][t]
            // repack h into bf16 A-fragment layout for next step
            int mt = b >> 4, M16 = b & 15;
            int kb = j >> 5, K = j & 31;
            int laneA = M16 + (((K >> 3) & 1) ? 16 : 0);
            int hh    = (K & 7) + ((K >= 16) ? 8 : 0);
            hnext[(((size_t)(mt * KBLK + kb)) * 32 + laneA) * 16 + hh] = (__bf16)hn;
            if (t == TSTEPS - 1) {
                size_t e = (size_t)b * HDIM + j;
                out[(size_t)BSZ * HDIM * TSTEPS + e] = hn;              // h_T
                out[(size_t)BSZ * HDIM * TSTEPS + BSZ * HDIM + e] = cn; // c_T
            }
        }
        // single grid barrier per step: publish h_t before anyone's GEMM(t+1)
        grid_sync(bar, gridDim.x);
    }
}

extern "C" void kernel_launch(void* const* d_in, const int* in_sizes, int n_in,
                              void* d_out, int out_size, void* d_ws, size_t ws_size,
                              hipStream_t stream) {
    const float* x    = (const float*)d_in[0];
    const float* U    = (const float*)d_in[1];
    const float* V    = (const float*)d_in[2];
    const float* bias = (const float*)d_in[3];

    char* ws = (char*)d_ws;
    __bf16*   ufrag  = (__bf16*)(ws + OFF_UFRAG);
    __bf16*   vfrag  = (__bf16*)(ws + OFF_VFRAG);
    __bf16*   xfrag  = (__bf16*)(ws + OFF_XFRAG);
    float*    gatesx = (float*)(ws + OFF_GATESX);
    __bf16*   hfrag  = (__bf16*)(ws + OFF_HFRAG);
    unsigned* bar    = (unsigned*)(ws + OFF_BAR);

    hipMemsetAsync(hfrag, 0, (size_t)2 * BSZ * HDIM * 2, stream);  // h ping-pong = 0
    hipMemsetAsync(bar,   0, 2 * sizeof(unsigned),       stream);

    pack_w<<<4096, 256, 0, stream>>>(U, ufrag);
    pack_w<<<4096, 256, 0, stream>>>(V, vfrag);
    pack_x<<<32768, 256, 0, stream>>>(x, xfrag);
    gemm_xu<<<16384, 256, 0, stream>>>(xfrag, ufrag, bias, gatesx);
    lstm_recurrent<<<64, 256, 73728, stream>>>(vfrag, gatesx, hfrag,
                                               (float*)d_out, bar);
}
